// SimRel_64862596104250
// MI455X (gfx1250) — matrix operand order repaired and checked by reading, not compiled, not run on hardware
//
#include <hip/hip_runtime.h>
#include <math.h>

// SimRel on MI455X (gfx1250, wave32).
// B=16, T=2048, D=768, K=64. Steps = 32768 split into 256 chunks of 128.
// d_in[0]: inputs f32 [32768][768]
// d_in[1]: labels int32 [32768]   (jax default x64 disabled -> int32)
// d_in[2]: class_avgs f32 [64][768]
// d_out  : sims f32 [32768][64]
// d_ws   : needs 256*64*768*4 B (48 MB) partial sums + 256*64*4 B counts.

#define EPSV 1e-8f
#define CHUNK 128
#define NCHUNK 256
#define KC 64
#define DD 768
#define ASTR 136          // staged-row stride in bf16 elems (128 + 8 pad)
#define NTILE 68          // 32 Gb tiles + 36 upper-triangular Gram tiles

typedef __attribute__((ext_vector_type(16))) __bf16 v16bf;
typedef __attribute__((ext_vector_type(8)))  __bf16 v8bf;
typedef __attribute__((ext_vector_type(8)))  float  v8f;

// Build a v16bf fragment from two 16B-aligned 8-element runs in LDS.
__device__ __forceinline__ v16bf frag16(const __bf16* p, int gap) {
  v8bf x = *(const v8bf*)p;
  v8bf y = *(const v8bf*)(p + gap);
  v16bf r;
#pragma unroll
  for (int e = 0; e < 8; ++e) { r[e] = x[e]; r[8 + e] = y[e]; }
  return r;
}

// Tile index -> (mt, nt, isGb). t<32: Gb 8x4. t>=32: Gram upper (mt<=nt) 36.
__device__ __forceinline__ void decodeTile(int t, int& mt, int& nt, bool& isGb) {
  if (t < 32) { isGb = true; mt = t >> 2; nt = t & 3; }
  else {
    isGb = false;
    int u = t - 32, m = 0;
    while (u >= 8 - m) { u -= 8 - m; ++m; }
    mt = m; nt = m + u;
  }
}

// ---------------- Phase 0: per-chunk class sums + counts ----------------
__global__ __launch_bounds__(256) void simrel_phase0(
    const float* __restrict__ inputs, const int* __restrict__ labels,
    float* __restrict__ partSums, int* __restrict__ partCnts) {
  __shared__ float ssum[KC * DD];   // 192 KB
  __shared__ int slab[CHUNK];
  const int c = blockIdx.x, tid = threadIdx.x;
  const size_t base = (size_t)c * CHUNK;
  for (int x = tid; x < KC * DD; x += 256) ssum[x] = 0.f;
  if (tid < CHUNK) slab[tid] = labels[base + tid];
  __syncthreads();
  // Thread t exclusively owns columns {t, t+256, t+512}: race-free LDS RMW.
  for (int i = 0; i < CHUNK; ++i) {
    const int lb = slab[i];
    const float* row = inputs + (base + i) * (size_t)DD;
#pragma unroll
    for (int q = 0; q < 3; ++q) {
      const int d = tid + q * 256;
      ssum[lb * DD + d] += row[d];
    }
  }
  __syncthreads();
  float* dst = partSums + (size_t)c * (KC * DD);
  for (int x = tid; x < KC * DD; x += 256) dst[x] = ssum[x];
  if (tid < KC) {
    int cnt = 0;
    for (int i = 0; i < CHUNK; ++i) cnt += (slab[i] == tid);
    partCnts[c * KC + tid] = cnt;
  }
}

// ------------- Phase 1: in-place exclusive prefix over chunks -----------
__global__ __launch_bounds__(256) void simrel_phase1(
    float* __restrict__ partSums, int* __restrict__ partCnts) {
  const int kd = blockIdx.x * blockDim.x + threadIdx.x;
  const size_t stride = (size_t)KC * DD;
  if (kd < KC * DD) {
    float run = 0.f;
    float* p = partSums + kd;
    for (int cc = 0; cc < NCHUNK; ++cc) {
      float t = p[(size_t)cc * stride];
      p[(size_t)cc * stride] = run;
      run += t;
    }
  }
  if (kd < KC) {
    int run = 0;
    int* q = partCnts + kd;
    for (int cc = 0; cc < NCHUNK; ++cc) {
      int t = q[cc * KC];
      q[cc * KC] = run;
      run += t;
    }
  }
}

// ------- Phase 2: WMMA GEMMs (Gb, Gram) + in-chunk scan + output --------
__global__ __launch_bounds__(256) void simrel_phase2(
    const float* __restrict__ inputs, const int* __restrict__ labels,
    const float* __restrict__ avgs, const float* __restrict__ baseSums,
    const int* __restrict__ baseCnts, float* __restrict__ out) {
  // Staged split-bf16 operand planes (converted once per element per block).
  __shared__ __attribute__((aligned(16))) __bf16 aHi[CHUNK * ASTR];
  __shared__ __attribute__((aligned(16))) __bf16 aLo[CHUNK * ASTR];
  __shared__ __attribute__((aligned(16))) __bf16 bHi[KC * ASTR];
  __shared__ __attribute__((aligned(16))) __bf16 bLo[KC * ASTR];
  __shared__ float sG[CHUNK][CHUNK + 1];   // Gram
  __shared__ float sGb[CHUNK][KC + 1];     // vec . base
  __shared__ float sgp[CHUNK][KC + 1];     // masked gram prefix
  __shared__ int   slab[CHUNK];
  __shared__ int   scb[KC];
  __shared__ float sbNS[KC];               // ||Base_k||^2 (or ||avgs_k||^2)
  __shared__ int   sinf[KC];
  __shared__ float tmpNS[4][KC];
  __shared__ int   tmpInf[4][KC];

  const int c = blockIdx.x, tid = threadIdx.x;
  const size_t chunkBase = (size_t)c * CHUNK;
  const float* vec = inputs + chunkBase * (size_t)DD;

  if (tid < CHUNK) slab[tid] = labels[chunkBase + tid];
  if (tid < KC) scb[tid] = baseCnts[c * KC + tid];
  __syncthreads();

  { // parallel base-row ||.||^2 and inf scan: thread (k = tid&63, q = tid>>6)
    const int k = tid & 63, q = tid >> 6;
    const int cb = scb[k];
    const float* row = (cb > 0) ? baseSums + ((size_t)c * KC + k) * DD
                                : avgs + (size_t)k * DD;
    float ss = 0.f;
    int anyinf = 0;
    for (int d = q * 192; d < q * 192 + 192; ++d) {
      float x = row[d];
      ss += x * x;
      anyinf |= (int)isinf(x);
    }
    tmpNS[q][k] = ss;
    tmpInf[q][k] = anyinf;
  }
  __syncthreads();
  if (tid < KC) {
    sbNS[tid] = tmpNS[0][tid] + tmpNS[1][tid] + tmpNS[2][tid] + tmpNS[3][tid];
    sinf[tid] = (scb[tid] == 0)
                    ? (tmpInf[0][tid] | tmpInf[1][tid] | tmpInf[2][tid] | tmpInf[3][tid])
                    : 0;
  }

  // --- WMMA GEMMs over six staged 128-wide K blocks ---
  const int wave = tid >> 5, lane = tid & 31;
  const int lm = lane & 15;
  const int hi8 = (lane & 16) ? 8 : 0;     // A-frag K sub-offset
  const int koff = (lane & 16) ? 16 : 0;   // unused: B frags also use hi8

  v8f acc[9];
#pragma unroll
  for (int ti = 0; ti < 9; ++ti) {
    v8f z = {0.f, 0.f, 0.f, 0.f, 0.f, 0.f, 0.f, 0.f};
    acc[ti] = z;
  }

  for (int blk = 0; blk < 6; ++blk) {
    const int kbase = blk * 128;
    // Stage: convert f32 -> (hi,lo) bf16 planes, once per element.
    for (int x = tid; x < CHUNK * 128; x += 256) {
      const int r = x >> 7, ci = x & 127;
      const float v = vec[(size_t)r * DD + kbase + ci];
      const __bf16 h = (__bf16)v;
      aHi[r * ASTR + ci] = h;
      aLo[r * ASTR + ci] = (__bf16)(v - (float)h);
    }
    for (int x = tid; x < KC * 128; x += 256) {
      const int r = x >> 7, ci = x & 127;
      const float* rowp = (scb[r] > 0) ? baseSums + ((size_t)c * KC + r) * DD
                                       : avgs + (size_t)r * DD;
      const float v = rowp[kbase + ci];
      const __bf16 h = (__bf16)v;
      bHi[r * ASTR + ci] = h;
      bLo[r * ASTR + ci] = (__bf16)(v - (float)h);
    }
    __syncthreads();

#pragma unroll
    for (int ti = 0; ti < 9; ++ti) {
      const int t = wave + ti * 8;
      if (t < NTILE) {
        int mt, nt; bool isGb;
        decodeTile(t, mt, nt, isGb);
        const int aOff = (mt * 16 + lm) * ASTR;
        const int bOff = (nt * 16 + lm) * ASTR;
        const __bf16* bph = isGb ? bHi : aHi;
        const __bf16* bpl = isGb ? bLo : aLo;
#pragma unroll
        for (int ks = 0; ks < 4; ++ks) {
          const int k0 = ks * 32;
          const v16bf ah = frag16(aHi + aOff + k0 + hi8, 16);
          const v16bf al = frag16(aLo + aOff + k0 + hi8, 16);
          const v16bf bh = frag16(bph + bOff + k0 + hi8, 16);
          const v16bf bl = frag16(bpl + bOff + k0 + hi8, 16);
          // bf16x3 error-corrected f32: hi*hi + hi*lo + lo*hi
          acc[ti] = __builtin_amdgcn_wmma_f32_16x16x32_bf16(
              false, ah, false, bh, (short)0, acc[ti], false, false);
          acc[ti] = __builtin_amdgcn_wmma_f32_16x16x32_bf16(
              false, ah, false, bl, (short)0, acc[ti], false, false);
          acc[ti] = __builtin_amdgcn_wmma_f32_16x16x32_bf16(
              false, al, false, bh, (short)0, acc[ti], false, false);
        }
      }
    }
    __syncthreads();   // before restaging next K block
  }

  // Write accumulators to LDS result arrays.
  {
    const int mbase = (lane & 16) ? 8 : 0;
#pragma unroll
    for (int ti = 0; ti < 9; ++ti) {
      const int t = wave + ti * 8;
      if (t < NTILE) {
        int mt, nt; bool isGb;
        decodeTile(t, mt, nt, isGb);
#pragma unroll
        for (int r = 0; r < 8; ++r) {
          const int mm = mt * 16 + mbase + r;
          if (isGb) sGb[mm][nt * 16 + lm] = acc[ti][r];
          else      sG[mm][nt * 16 + lm] = acc[ti][r];
        }
      }
    }
  }
  __syncthreads();

  // Mirror lower-triangular Gram tiles (symmetry).
  for (int x = tid; x < CHUNK * CHUNK; x += 256) {
    const int i = x >> 7, j = x & 127;
    if ((i >> 4) > (j >> 4)) sG[i][j] = sG[j][i];
  }
  __syncthreads();

  // --- masked Gram prefix: gp[i][k] = sum_{j<i, lab_j==k} G[i][j] ---
  if (tid < CHUNK) {
    const int i = tid;
    for (int k2 = 0; k2 < KC; ++k2) sgp[i][k2] = 0.f;
    for (int j = 0; j < i; ++j) sgp[i][slab[j]] += sG[i][j];
  }
  __syncthreads();

  // --- per-class sequential walk (the only true dependency chain) ---
  if (tid < KC) {
    const int k = tid;
    const int cb = scb[k];
    const int isinfk = sinf[k];
    const float an_unseen = fmaxf(sqrtf(sbNS[k]), EPSV);
    float ns = (cb > 0) ? sbNS[k] : 0.f;   // ||S_k||^2 running
    int ci = 0;                             // in-chunk count of class k
    for (int i = 0; i < CHUNK; ++i) {
      const int lb = slab[i];
      const float gii = sG[i][i];
      const float vn = fmaxf(sqrtf(gii), EPSV);
      const float gb = sGb[i][k];
      const float gpv = sgp[i][k];
      float sim;
      if (cb == 0 && ci == 0) {
        // class average is still the initial class_avgs row
        sim = isinfk ? ((lb == k) ? 1.f : -1.f) : gb / (an_unseen * vn);
      } else {
        const float ctot = (float)(cb + ci);
        const float num = gpv + ((cb > 0) ? gb : 0.f);
        const float an = fmaxf(sqrtf(ns) / ctot, EPSV);
        sim = (num / ctot) / (an * vn);
      }
      out[(chunkBase + i) * (size_t)KC + k] = sim;
      if (lb == k) {                        // S_k += vec_i
        if (cb == 0 && ci == 0) ns = gii;   // first update discards initial row
        else ns += 2.f * (gpv + ((cb > 0) ? gb : 0.f)) + gii;
        ++ci;
      }
    }
  }
}

extern "C" void kernel_launch(void* const* d_in, const int* in_sizes, int n_in,
                              void* d_out, int out_size, void* d_ws, size_t ws_size,
                              hipStream_t stream) {
  const float* inputs = (const float*)d_in[0];
  const int* labels = (const int*)d_in[1];
  const float* avgs = (const float*)d_in[2];
  float* out = (float*)d_out;
  // workspace: [NCHUNK][KC][DD] f32 partial/prefix sums, then [NCHUNK][KC] int counts
  float* partSums = (float*)d_ws;
  int* partCnts = (int*)((char*)d_ws + (size_t)NCHUNK * KC * DD * sizeof(float));
  (void)in_sizes; (void)n_in; (void)out_size; (void)ws_size;

  simrel_phase0<<<NCHUNK, 256, 0, stream>>>(inputs, labels, partSums, partCnts);
  simrel_phase1<<<(KC * DD + 255) / 256, 256, 0, stream>>>(partSums, partCnts);
  simrel_phase2<<<NCHUNK, 256, 0, stream>>>(inputs, labels, avgs, partSums,
                                            partCnts, out);
}
